// CausalSelfAttention_57664230916457
// MI455X (gfx1250) — matrix operand order repaired and checked by reading, not compiled, not run on hardware
//
#include <hip/hip_runtime.h>

// ---------------------------------------------------------------------------
// CDNA5 (gfx1250) causal self-attention pipeline, bf16 WMMA + TDM async tiles.
//   1) cvt x, qkv_w, out_w  fp32 -> bf16
//   2) GEMM1 (NT, 128x128 tile, 8 waves, TDM double-buffered LDS tiles):
//      qkv = x @ qkv_w^T + b, epilogue scatters bf16 Q/K/V into [B,H,T,128]
//   3) RoPE in-place on Q,K (pairwise, race-free)
//   4) Flash attention: 16-query tile per wave, TDM-loaded K tile + manual
//      V^T tile, online softmax, P bounced through LDS (C->A layout)
//   5) GEMM2 (NT): out = attn @ out_w^T + b  (fp32 output)
// ---------------------------------------------------------------------------

typedef __bf16 bf16;
typedef __attribute__((ext_vector_type(4)))  __bf16 bf16x4;
typedef __attribute__((ext_vector_type(8)))  __bf16 bf16x8;
typedef __attribute__((ext_vector_type(16))) __bf16 v16bf;
typedef __attribute__((ext_vector_type(8)))  float  v8f;
typedef __attribute__((ext_vector_type(4)))  unsigned int u32x4;
typedef __attribute__((ext_vector_type(8)))  int i32x8;
typedef __attribute__((ext_vector_type(4)))  int i32x4;

union Frag { v16bf v; bf16x8 h[2]; };

#define WMMA_BF16(a, b, c) \
  __builtin_amdgcn_wmma_f32_16x16x32_bf16(false, (a), false, (b), (short)0, (c), false, false)

constexpr int Bn = 2, T = 2048, D = 2048, H = 16, HD = 128;

// ------------------------- TDM 2-D tile load -> LDS ------------------------
// D# per CDNA5 ISA ch.8: group0 = {flags, lds_addr, global_addr, type=2},
// group1 = {dsize/pad, tensor dims, tile dims, stride}. bf16 (data_size=1).
// pad_interval: (1<<v)*8B per row before padding; pad_amount: (v+1) DWORDs.
__device__ __forceinline__ void tdm_load_2d_bf16(
    unsigned lds_off, const void* gptr,
    unsigned tile_x_elems, unsigned tile_rows,
    unsigned tensor_d0, unsigned tensor_d1,
    unsigned long long row_stride_elems,
    int pad_interval, int pad_amount) {
  const unsigned long long ga = (unsigned long long)gptr;
  u32x4 g0;
  g0[0] = 1u;                                         // count=1, user descriptor
  g0[1] = lds_off;                                    // D#.lds_addr
  g0[2] = (unsigned)(ga & 0xffffffffu);               // global_addr[31:0]
  g0[3] = (unsigned)((ga >> 32) & 0x1ffffffu)         // global_addr[56:32]
        | (2u << 30);                                 // type = 2 ("image")
  unsigned d0 = (1u << 16);                           // data_size = 1 (2 bytes)
  if (pad_amount >= 0)
    d0 |= (1u << 20) | ((unsigned)pad_interval << 22) | ((unsigned)pad_amount << 25);
  i32x8 g1;
  g1[0] = (int)d0;
  g1[1] = (int)((tensor_d0 & 0xffffu) << 16);                         // dim0[15:0]
  g1[2] = (int)((tensor_d0 >> 16) | ((tensor_d1 & 0xffffu) << 16));   // dim0 hi | dim1 lo
  g1[3] = (int)((tensor_d1 >> 16) | (tile_x_elems << 16));            // dim1 hi | tile_dim0
  g1[4] = (int)(tile_rows & 0xffffu);                                 // tile_dim1 (dim2=0)
  g1[5] = (int)(row_stride_elems & 0xffffffffu);                      // dim0_stride lo
  g1[6] = (int)((row_stride_elems >> 32) & 0xffffu);                  // stride hi (dim1_stride=0)
  g1[7] = 0;
  const i32x4 z4 = {0, 0, 0, 0};
#if __clang_major__ >= 23
  const i32x8 z8 = {0, 0, 0, 0, 0, 0, 0, 0};
  __builtin_amdgcn_tensor_load_to_lds(g0, g1, z4, z4, z8, 0);
#else
  __builtin_amdgcn_tensor_load_to_lds(g0, g1, z4, z4, 0);
#endif
}

__device__ __forceinline__ unsigned lds_off_u32(const void* p) {
  // generic LDS pointer keeps the LDS byte offset in addr[31:0] (aperture rule)
  return (unsigned)(unsigned long long)p;
}

// ------------------------------ fp32 -> bf16 -------------------------------
__global__ __launch_bounds__(256) void cvt_f32_to_bf16(
    const float* __restrict__ s, bf16* __restrict__ d, int n4) {
  int i = blockIdx.x * 256 + threadIdx.x;
  if (i >= n4) return;
  float4 f = ((const float4*)s)[i];
  bf16x4 o;
  o[0] = (bf16)f.x; o[1] = (bf16)f.y; o[2] = (bf16)f.z; o[3] = (bf16)f.w;
  ((bf16x4*)d)[i] = o;
}

// ------------------------------ NT bf16 GEMM -------------------------------
// C[m][n] = sum_k A[m][k] * Bw[n][k] + bias[n]
// 128x128 C tile per block; 8 waves in 2(M)x4(N); each wave 64x32 = 4x2 WMMA.
// K-slabs of 32 streamed by TDM into double-buffered padded LDS tiles.
template <int MODE>   // 0: scatter bf16 QKV, 1: fp32 dense out
__global__ __launch_bounds__(256) void gemm_bf16_nt(
    const bf16* __restrict__ A, const bf16* __restrict__ Bw,
    const float* __restrict__ bias, float* __restrict__ Cf,
    bf16* __restrict__ Qb, bf16* __restrict__ Kb, bf16* __restrict__ Vb,
    int K) {
  __shared__ bf16 sA[2][128][40];   // 32 k + 8 pad (row stride 80B = TDM pad 3/3)
  __shared__ bf16 sB[2][128][40];

  const int tid  = threadIdx.x;
  const int lane = tid & 31, wave = tid >> 5;
  const int wm = wave >> 2, wn = wave & 3;
  const int half = lane >> 4, l16 = lane & 15;
  const int m0 = blockIdx.y * 128, n0 = blockIdx.x * 128;
  const bool issuer = (wave == 0);

  v8f acc[4][2] = {};

  const int nk = K >> 5;
  if (issuer) {   // prologue tile 0
    tdm_load_2d_bf16(lds_off_u32(&sA[0][0][0]), A  + (size_t)m0 * K, 32, 128,
                     (unsigned)K, 1u << 20, (unsigned)K, 3, 3);
    tdm_load_2d_bf16(lds_off_u32(&sB[0][0][0]), Bw + (size_t)n0 * K, 32, 128,
                     (unsigned)K, 1u << 20, (unsigned)K, 3, 3);
  }

  for (int ki = 0; ki < nk; ++ki) {
    const int buf = ki & 1;
    if (issuer) __builtin_amdgcn_s_wait_tensorcnt(0);
    __syncthreads();                         // tile[buf] ready for everyone
    if (issuer && ki + 1 < nk) {             // prefetch next slab into other buf
      const int k1 = (ki + 1) << 5;
      tdm_load_2d_bf16(lds_off_u32(&sA[buf ^ 1][0][0]), A  + (size_t)m0 * K + k1,
                       32, 128, (unsigned)K, 1u << 20, (unsigned)K, 3, 3);
      tdm_load_2d_bf16(lds_off_u32(&sB[buf ^ 1][0][0]), Bw + (size_t)n0 * K + k1,
                       32, 128, (unsigned)K, 1u << 20, (unsigned)K, 3, 3);
    }

    Frag a[4], b[2];
#pragma unroll
    for (int mi = 0; mi < 4; ++mi) {             // A frag: lanes 0-15 K 0-7/16-23
      const int r = wm * 64 + mi * 16 + l16;
      a[mi].h[0] = *(const bf16x8*)&sA[buf][r][half * 8];
      a[mi].h[1] = *(const bf16x8*)&sA[buf][r][half * 8 + 16];
    }
#pragma unroll
    for (int ni = 0; ni < 2; ++ni) {             // B frag: lanes 0-15 K 0-7/16-23
      const int c = wn * 32 + ni * 16 + l16;
      b[ni].h[0] = *(const bf16x8*)&sB[buf][c][half * 8];
      b[ni].h[1] = *(const bf16x8*)&sB[buf][c][half * 8 + 16];
    }
#pragma unroll
    for (int mi = 0; mi < 4; ++mi)
#pragma unroll
      for (int ni = 0; ni < 2; ++ni)
        acc[mi][ni] = WMMA_BF16(a[mi].v, b[ni].v, acc[mi][ni]);
    __syncthreads();                         // done reading tile[buf]
  }

#pragma unroll
  for (int mi = 0; mi < 4; ++mi) {
#pragma unroll
    for (int ni = 0; ni < 2; ++ni) {
      const int n  = n0 + wn * 32 + ni * 16 + l16;
      const float bv = bias[n];
#pragma unroll
      for (int r = 0; r < 8; ++r) {
        const int m  = m0 + wm * 64 + mi * 16 + half * 8 + r;
        const float val = acc[mi][ni][r] + bv;
        if (MODE == 0) {
          const int sIdx = n >> 11;           // 0:q 1:k 2:v
          const int h    = (n >> 7) & 15;
          const int dd   = n & 127;
          const int bb   = m >> 11;
          const int t    = m & 2047;
          bf16* dst = (sIdx == 0) ? Qb : (sIdx == 1) ? Kb : Vb;
          dst[(((size_t)(bb * H + h)) * T + t) * HD + dd] = (bf16)val;
        } else {
          Cf[(size_t)m * D + n] = val;
        }
      }
    }
  }
}

// ------------------------------- RoPE in-place -----------------------------
__global__ __launch_bounds__(256) void rope_inplace(
    bf16* __restrict__ Qb, bf16* __restrict__ Kb,
    const float* __restrict__ ctab, const float* __restrict__ stab) {
  const size_t idx = (size_t)blockIdx.x * 256 + threadIdx.x;  // Bn*H*T*64
  const int    dd  = (int)(idx & 63);
  const size_t row = idx >> 6;
  const int    t   = (int)(row & (T - 1));
  const float c0 = ctab[t * HD + dd],      s0 = stab[t * HD + dd];
  const float c1 = ctab[t * HD + dd + 64], s1 = stab[t * HD + dd + 64];
  bf16* bufs[2] = {Qb, Kb};
#pragma unroll
  for (int i = 0; i < 2; ++i) {
    bf16* p = bufs[i] + row * HD;
    const float x0 = (float)p[dd], x1 = (float)p[dd + 64];
    p[dd]      = (bf16)(x0 * c0 - x1 * s0);
    p[dd + 64] = (bf16)(x1 * c1 + x0 * s1);
  }
}

// ------------------------------ flash attention ----------------------------
// grid (16, B*H); 8 waves/block, wave w handles queries [qblk*128+16w, +16).
__global__ __launch_bounds__(256) void flash_attn(
    const bf16* __restrict__ Qb, const bf16* __restrict__ Kb,
    const bf16* __restrict__ Vb, bf16* __restrict__ AO) {
  __shared__ bf16 sK[32][136];      // keys x dims (272B rows = TDM pad 5/3)
  __shared__ bf16 sVt[128][40];     // dims x keys
  __shared__ bf16 sP[8][16][40];    // per-wave C->A bounce

  const int tid  = threadIdx.x;
  const int lane = tid & 31, wave = tid >> 5;
  const int half = lane >> 4, l16 = lane & 15;
  const int qblk = blockIdx.x;
  const int bh   = blockIdx.y;
  const int qbase = qblk * 128 + wave * 16;
  const bool issuer = (wave == 0);

  Frag aq[4];                        // Q 16x128 as four 16x32 A fragments
  {
    const bf16* qp = Qb + ((size_t)bh * T + qbase + l16) * HD;
#pragma unroll
    for (int c = 0; c < 4; ++c) {
      aq[c].h[0] = *(const bf16x8*)(qp + c * 32 + half * 8);
      aq[c].h[1] = *(const bf16x8*)(qp + c * 32 + half * 8 + 16);
    }
  }

  v8f o[8] = {};
  float mrun[8], lrun[8];
#pragma unroll
  for (int r = 0; r < 8; ++r) { mrun[r] = -3.0e38f; lrun[r] = 0.f; }

  const int row_l = tid >> 4;            // 0..15 (two passes -> 32 key rows)
  const int ckd   = (tid & 15) * 8;      // dim chunk
  const int nkt   = qblk * 4 + 4;        // causal key-tile count for block
  constexpr float SCALE = 0.08838834764831845f;  // 1/sqrt(128)

  for (int kt = 0; kt < nkt; ++kt) {
    const int kb = kt * 32;
    __syncthreads();                     // previous S/PV reads complete
    if (issuer)                          // TDM: K tile 32 keys x 128 dims
      tdm_load_2d_bf16(lds_off_u32(&sK[0][0]),
                       Kb + ((size_t)bh * T + kb) * HD,
                       128, 32, 128u, (unsigned)T, 128ull, 5, 3);
#pragma unroll
    for (int rr = 0; rr < 2; ++rr) {     // manual V^T tile fill (transpose)
      const int row = row_l + rr * 16;
      const bf16x8 vv = *(const bf16x8*)(Vb + ((size_t)bh * T + kb + row) * HD + ckd);
#pragma unroll
      for (int e = 0; e < 8; ++e) sVt[ckd + e][row] = vv[e];
    }
    if (issuer) __builtin_amdgcn_s_wait_tensorcnt(0);
    __syncthreads();

    if (kb <= qbase + 15) {              // wave-uniform causal skip
      v8f s[2] = {};
#pragma unroll
      for (int j = 0; j < 2; ++j) {      // S = Q K^T  (two 16-key subtiles)
#pragma unroll
        for (int c = 0; c < 4; ++c) {
          Frag bk;
          bk.h[0] = *(const bf16x8*)&sK[j * 16 + l16][c * 32 + half * 8];
          bk.h[1] = *(const bf16x8*)&sK[j * 16 + l16][c * 32 + half * 8 + 16];
          s[j] = WMMA_BF16(aq[c].v, bk.v, s[j]);
        }
      }
      const int qrow0 = qbase + half * 8;
#pragma unroll
      for (int r = 0; r < 8; ++r) {      // online softmax per row
        const int q = qrow0 + r;
        float s0 = s[0][r] * SCALE, s1 = s[1][r] * SCALE;
        if (kb + l16 > q)      s0 = -3.0e38f;
        if (kb + 16 + l16 > q) s1 = -3.0e38f;
        float tmax = fmaxf(s0, s1);
#pragma unroll
        for (int off = 1; off < 16; off <<= 1)
          tmax = fmaxf(tmax, __shfl_xor(tmax, off, 32));
        const float mnew  = fmaxf(mrun[r], tmax);
        const float alpha = __expf(mrun[r] - mnew);
        const float p0 = __expf(s0 - mnew), p1 = __expf(s1 - mnew);
        float rs = p0 + p1;
#pragma unroll
        for (int off = 1; off < 16; off <<= 1)
          rs += __shfl_xor(rs, off, 32);
        lrun[r] = lrun[r] * alpha + rs;
        mrun[r] = mnew;
#pragma unroll
        for (int nn = 0; nn < 8; ++nn) o[nn][r] *= alpha;
        sP[wave][half * 8 + r][l16]      = (bf16)p0;   // C-layout -> LDS
        sP[wave][half * 8 + r][16 + l16] = (bf16)p1;
      }
      asm volatile("s_wait_dscnt 0" ::: "memory");     // wave-private bounce
      Frag ap;                                          // read back A-layout
      ap.h[0] = *(const bf16x8*)&sP[wave][l16][half * 8];
      ap.h[1] = *(const bf16x8*)&sP[wave][l16][half * 8 + 16];
#pragma unroll
      for (int nn = 0; nn < 8; ++nn) {   // O += P V
        Frag bv;
        bv.h[0] = *(const bf16x8*)&sVt[nn * 16 + l16][half * 8];
        bv.h[1] = *(const bf16x8*)&sVt[nn * 16 + l16][half * 8 + 16];
        o[nn] = WMMA_BF16(ap.v, bv.v, o[nn]);
      }
    }
  }

  const int b = bh >> 4, h = bh & 15;    // write bf16 straight into [B,T,D]
#pragma unroll
  for (int r = 0; r < 8; ++r) {
    const float inv = 1.0f / lrun[r];
    const size_t base = ((size_t)b * T + qbase + half * 8 + r) * D + h * HD;
#pragma unroll
    for (int nn = 0; nn < 8; ++nn)
      AO[base + nn * 16 + l16] = (bf16)(o[nn][r] * inv);
  }
}

// --------------------------------- launch ----------------------------------
extern "C" void kernel_launch(void* const* d_in, const int* in_sizes, int n_in,
                              void* d_out, int out_size, void* d_ws, size_t ws_size,
                              hipStream_t stream) {
  (void)in_sizes; (void)n_in; (void)out_size; (void)ws_size;
  const float* x    = (const float*)d_in[0];
  const float* cosT = (const float*)d_in[1];
  const float* sinT = (const float*)d_in[2];
  const float* qkvw = (const float*)d_in[3];
  const float* qkvb = (const float*)d_in[4];
  const float* outw = (const float*)d_in[5];
  const float* outb = (const float*)d_in[6];
  float* out = (float*)d_out;

  char* ws = (char*)d_ws;
  size_t off = 0;
  auto alloc = [&](size_t bytes) -> void* {
    void* p = ws + off;
    off += (bytes + 255) & ~(size_t)255;
    return p;
  };
  bf16* Xb = (bf16*)alloc((size_t)Bn * T * D * 2);        // 16 MB
  bf16* Wq = (bf16*)alloc((size_t)3 * D * D * 2);         // 24 MB
  bf16* Wo = (bf16*)alloc((size_t)D * D * 2);             //  8 MB
  bf16* Qb = (bf16*)alloc((size_t)Bn * H * T * HD * 2);   // 16 MB
  bf16* Kb = (bf16*)alloc((size_t)Bn * H * T * HD * 2);   // 16 MB
  bf16* Vb = (bf16*)alloc((size_t)Bn * H * T * HD * 2);   // 16 MB
  bf16* AO = (bf16*)alloc((size_t)Bn * T * D * 2);        // 16 MB  (~112 MB total)

  cvt_f32_to_bf16<<<(Bn * T * D / 4 + 255) / 256, 256, 0, stream>>>(x, Xb, Bn * T * D / 4);
  cvt_f32_to_bf16<<<(3 * D * D / 4 + 255) / 256, 256, 0, stream>>>(qkvw, Wq, 3 * D * D / 4);
  cvt_f32_to_bf16<<<(D * D / 4 + 255) / 256, 256, 0, stream>>>(outw, Wo, D * D / 4);

  gemm_bf16_nt<0><<<dim3(3 * D / 128, Bn * T / 128), 256, 0, stream>>>(
      Xb, Wq, qkvb, nullptr, Qb, Kb, Vb, D);

  rope_inplace<<<(Bn * H * T * 64) / 256, 256, 0, stream>>>(Qb, Kb, cosT, sinT);

  flash_attn<<<dim3(T / 128, Bn * H), 256, 0, stream>>>(Qb, Kb, Vb, AO);

  gemm_bf16_nt<1><<<dim3(D / 128, Bn * T / 128), 256, 0, stream>>>(
      AO, Wo, outb, out, nullptr, nullptr, nullptr, D);
}
